// MoeLayer_29291676958946
// MI455X (gfx1250) — matrix operand order repaired and checked by reading, not compiled, not run on hardware
//
#include <hip/hip_runtime.h>

typedef __attribute__((ext_vector_type(16))) __bf16 v16bf;
typedef __attribute__((ext_vector_type(8)))  __bf16 v8bf;
typedef __attribute__((ext_vector_type(8)))  float  v8f;
typedef __attribute__((ext_vector_type(4)))  float  v4f;
typedef __attribute__((ext_vector_type(4)))  int    v4i;

constexpr int E     = 8;
constexpr int D_IN  = 1024;
constexpr int D_OUT = 1024;
constexpr int NTOK  = 8192;

constexpr int BM = 128;  // rows (tokens) per block
constexpr int BN = 32;   // output cols per block
constexpr int BK = 32;   // K per step (one bf16 WMMA K)
constexpr int LDS_STRIDE = 40; // bf16/row (80B padded: conflict-free, 16B aligned)

#if defined(__has_builtin)
#  if __has_builtin(__builtin_amdgcn_global_load_async_to_lds_b128) && \
      __has_builtin(__builtin_amdgcn_s_wait_asynccnt)
#    define USE_ASYNC_LDS 1
#  endif
#endif
#ifndef USE_ASYNC_LDS
#  define USE_ASYNC_LDS 0
#endif

typedef __attribute__((address_space(1))) v4i gv4i;  // global
typedef __attribute__((address_space(3))) v4i lv4i;  // LDS

// ---------------------------------------------------------------------------
// Kernel 1: gate logits + softmax.  One wave (32 lanes) per token.
// ---------------------------------------------------------------------------
__global__ __launch_bounds__(256) void gate_kernel(const float* __restrict__ x,
                                                   const float* __restrict__ Wg,
                                                   const float* __restrict__ bg,
                                                   float* __restrict__ g)
{
    const int wave = threadIdx.x >> 5;
    const int lane = threadIdx.x & 31;
    const int n = blockIdx.x * 8 + wave;

    const float* xr = x + (size_t)n * D_IN;
    float acc[E];
#pragma unroll
    for (int e = 0; e < E; ++e) acc[e] = 0.0f;

    for (int i = lane; i < D_IN; i += 32) {
        const float xv = xr[i];
        const float* wr = Wg + (size_t)i * E;
#pragma unroll
        for (int e = 0; e < E; ++e) acc[e] = fmaf(xv, wr[e], acc[e]);
    }
#pragma unroll
    for (int e = 0; e < E; ++e) {
        for (int off = 16; off > 0; off >>= 1)
            acc[e] += __shfl_xor(acc[e], off, 32);
    }
#pragma unroll
    for (int e = 0; e < E; ++e) acc[e] += bg[e];

    float m = acc[0];
#pragma unroll
    for (int e = 1; e < E; ++e) m = fmaxf(m, acc[e]);
    float s = 0.0f;
#pragma unroll
    for (int e = 0; e < E; ++e) { acc[e] = __expf(acc[e] - m); s += acc[e]; }
    const float inv = 1.0f / s;

    if (lane == 0) {
        float4 g0 = make_float4(acc[0]*inv, acc[1]*inv, acc[2]*inv, acc[3]*inv);
        float4 g1 = make_float4(acc[4]*inv, acc[5]*inv, acc[6]*inv, acc[7]*inv);
        *(float4*)(g + (size_t)n * E)     = g0;
        *(float4*)(g + (size_t)n * E + 4) = g1;
    }
}

// ---------------------------------------------------------------------------
// Kernel 2: We [e][i][o] f32 -> WebT [e][o][i] bf16  (LDS-tiled transpose)
// ---------------------------------------------------------------------------
__global__ __launch_bounds__(256) void webt_kernel(const float* __restrict__ We,
                                                   __bf16* __restrict__ WebT)
{
    __shared__ __bf16 tile[32][36];
    const int e  = blockIdx.z;
    const int i0 = blockIdx.y * 32;
    const int o0 = blockIdx.x * 32;
    const int r  = threadIdx.x >> 3;
    const int c4 = (threadIdx.x & 7) * 4;

    const float* src = We + ((size_t)e * D_IN + (i0 + r)) * D_OUT + o0 + c4;
    v4f v = *(const v4f*)src;
    tile[r][c4 + 0] = (__bf16)v[0];
    tile[r][c4 + 1] = (__bf16)v[1];
    tile[r][c4 + 2] = (__bf16)v[2];
    tile[r][c4 + 3] = (__bf16)v[3];
    __syncthreads();

    __bf16* dst = WebT + ((size_t)e * D_OUT + (o0 + r)) * D_IN + i0 + c4;
    dst[0] = tile[c4 + 0][r];
    dst[1] = tile[c4 + 1][r];
    dst[2] = tile[c4 + 2][r];
    dst[3] = tile[c4 + 3][r];
}

// ---------------------------------------------------------------------------
// Kernel 3: fused 8-expert GEMM + gate-weighted combine + bias.
// 256 thr = 8 waves. Block tile 128x32; wave owns 32 rows x 16 cols,
// one f32x8 accumulator per m-subtile per expert. Double-buffered LDS;
// B tiles staged with GLOBAL_LOAD_ASYNC_TO_LDS_B128; B fragments
// software-pipelined one expert ahead of their WMMAs.
// ---------------------------------------------------------------------------
__global__ __launch_bounds__(256) void moe_gemm_kernel(
    const float*  __restrict__ x,     // [N, D_IN] f32
    const __bf16* __restrict__ WebT,  // [E, D_OUT, D_IN] bf16
    const float*  __restrict__ be,    // [E, D_OUT] f32
    const float*  __restrict__ g,     // [N, E] f32
    float*        __restrict__ out)   // [N, D_OUT] f32
{
    __shared__ __attribute__((aligned(16))) __bf16 As[2][BM][LDS_STRIDE];
    __shared__ __attribute__((aligned(16))) __bf16 Bs[2][E][BN][LDS_STRIDE];

    const int t    = threadIdx.x;
    const int wave = t >> 5;
    const int lane = t & 31;
    const int wm   = wave & 3;   // 0..3 : 32-row sub-tile
    const int wn   = wave >> 2;  // 0..1 : 16-col sub-tile
    const int bm   = blockIdx.x * BM;
    const int bn   = blockIdx.y * BN;

    v8f acc[2][E];
#pragma unroll
    for (int mt = 0; mt < 2; ++mt)
#pragma unroll
        for (int e = 0; e < E; ++e)
            acc[mt][e] = (v8f){0.f,0.f,0.f,0.f,0.f,0.f,0.f,0.f};

    // ---- A staging role: thread covers 16 f32 of one row ----
    const int am = t >> 1;            // 0..127
    const int ak = (t & 1) * 16;      // 0 or 16
    const float* aptr = x + (size_t)(bm + am) * D_IN + ak;

    // ---- B staging role: wave == expert; 4 lanes cover one 64B k-row ----
    const int bE    = wave;           // expert 0..7
    const int brow0 = lane >> 2;      // 0..7 (j adds 8,16,24)
    const int bcol  = (lane & 3) * 8; // bf16 offset within row (16B chunks)

    auto stageA = [&](int buf, int kb) {
        v4f a0 = *(const v4f*)(aptr + kb);
        v4f a1 = *(const v4f*)(aptr + kb + 4);
        v4f a2 = *(const v4f*)(aptr + kb + 8);
        v4f a3 = *(const v4f*)(aptr + kb + 12);
        v8bf v0, v1;
#pragma unroll
        for (int i = 0; i < 4; ++i) {
            v0[i] = (__bf16)a0[i]; v0[i + 4] = (__bf16)a1[i];
            v1[i] = (__bf16)a2[i]; v1[i + 4] = (__bf16)a3[i];
        }
        *(v8bf*)&As[buf][am][ak]     = v0;
        *(v8bf*)&As[buf][am][ak + 8] = v1;
    };

    auto stageB = [&](int buf, int kb) {
#pragma unroll
        for (int j = 0; j < 4; ++j) {
            const int n = brow0 + j * 8;
            const __bf16* s = WebT + ((size_t)bE * D_OUT + (bn + n)) * D_IN + kb + bcol;
            __bf16* d = &Bs[buf][bE][n][bcol];
#if USE_ASYNC_LDS
            __builtin_amdgcn_global_load_async_to_lds_b128(
                (gv4i*)s, (lv4i*)d, 0, 0);
#else
            *(v8bf*)d = *(const v8bf*)s;
#endif
        }
    };

    // fragment addressing (ISA 7.12.2 layouts)
    const int frow  = lane & 15;
    const int koffA = (lane < 16) ? 0 : 8;
    const int koffB = (lane < 16) ? 0 : 8;

    auto loadB = [&](int buf, int e) -> v16bf {
        v8bf blo = *(const v8bf*)&Bs[buf][e][wn * 16 + frow][koffB];
        v8bf bhi = *(const v8bf*)&Bs[buf][e][wn * 16 + frow][koffB + 16];
        v16bf b;
#pragma unroll
        for (int i = 0; i < 8; ++i) { b[i] = blo[i]; b[i + 8] = bhi[i]; }
        return b;
    };

    // prologue
    stageA(0, 0);
    stageB(0, 0);

    int buf = 0;
    for (int kb = 0; kb < D_IN; kb += BK) {
#if USE_ASYNC_LDS
        __builtin_amdgcn_s_wait_asynccnt(0);
#endif
        __syncthreads();
        const int nkb = kb + BK;
        if (nkb < D_IN) { stageA(buf ^ 1, nkb); stageB(buf ^ 1, nkb); }

        // A fragments (two 16-row sub-tiles)
        v16bf af[2];
#pragma unroll
        for (int mt = 0; mt < 2; ++mt) {
            v8bf alo = *(const v8bf*)&As[buf][wm * 32 + mt * 16 + frow][koffA];
            v8bf ahi = *(const v8bf*)&As[buf][wm * 32 + mt * 16 + frow][koffA + 16];
#pragma unroll
            for (int i = 0; i < 8; ++i) { af[mt][i] = alo[i]; af[mt][i + 8] = ahi[i]; }
        }

        // 16 WMMAs: 8 experts x 2 m-subtiles; B frags pipelined 1 ahead
        v16bf bcur = loadB(buf, 0);
#pragma unroll
        for (int e = 0; e < E; ++e) {
            v16bf bnext = bcur;
            if (e + 1 < E) bnext = loadB(buf, e + 1);
            acc[0][e] = __builtin_amdgcn_wmma_f32_16x16x32_bf16(
                false, af[0], false, bcur, (short)0, acc[0][e], false, false);
            acc[1][e] = __builtin_amdgcn_wmma_f32_16x16x32_bf16(
                false, af[1], false, bcur, (short)0, acc[1][e], false, false);
            bcur = bnext;
        }
        buf ^= 1;
    }

    // ---- epilogue: out[n,o] = sum_e g[n,e] * (acc_e + be[e,o]) ----
    const int col = bn + wn * 16 + (lane & 15);
    float bev[E];
#pragma unroll
    for (int e = 0; e < E; ++e) bev[e] = be[(size_t)e * D_OUT + col];

#pragma unroll
    for (int mt = 0; mt < 2; ++mt) {
        const int rbase = bm + wm * 32 + mt * 16 + ((lane < 16) ? 0 : 8);
#pragma unroll
        for (int r = 0; r < 8; ++r) {
            const int row = rbase + r;
            const float* gr = g + (size_t)row * E;
            v4f g0 = *(const v4f*)gr;
            v4f g1 = *(const v4f*)(gr + 4);
            float s = 0.0f;
#pragma unroll
            for (int e = 0; e < 4; ++e) s += g0[e] * (acc[mt][e][r] + bev[e]);
#pragma unroll
            for (int e = 0; e < 4; ++e) s += g1[e] * (acc[mt][e + 4][r] + bev[e + 4]);
            out[(size_t)row * D_OUT + col] = s;
        }
    }
}

// ---------------------------------------------------------------------------
extern "C" void kernel_launch(void* const* d_in, const int* in_sizes, int n_in,
                              void* d_out, int out_size, void* d_ws, size_t ws_size,
                              hipStream_t stream)
{
    const float* x  = (const float*)d_in[0];  // [N, D_IN]
    const float* We = (const float*)d_in[1];  // [E, D_IN, D_OUT]
    const float* be = (const float*)d_in[2];  // [E, D_OUT]
    const float* Wg = (const float*)d_in[3];  // [D_IN, E]
    const float* bg = (const float*)d_in[4];  // [E]
    float* out = (float*)d_out;               // [N, D_OUT]

    float*  g    = (float*)d_ws;                                        // 256 KB
    __bf16* WebT = (__bf16*)((char*)d_ws + (size_t)NTOK * E * sizeof(float)); // 16 MB

    gate_kernel<<<NTOK / 8, 256, 0, stream>>>(x, Wg, bg, g);
    webt_kernel<<<dim3(D_OUT / 32, D_IN / 32, E), 256, 0, stream>>>(We, WebT);
    moe_gemm_kernel<<<dim3(NTOK / BM, D_OUT / BN), 256, 0, stream>>>(x, WebT, be, g, out);
}
